// CrossAttentionLayerExport_39359080300921
// MI455X (gfx1250) — matrix operand order repaired and checked by reading, not compiled, not run on hardware
//
#include <hip/hip_runtime.h>

#define HID 256
#define NHEADS 8
#define HDIM 32
#define RBFD 50
#define LSZ 1024
#define PSZ 2048
#define PCHUNKS 4
#define PCH (PSZ / PCHUNKS)  // 512 proteins per attend block

typedef __attribute__((ext_vector_type(16))) __bf16 v16bf;
typedef __attribute__((ext_vector_type(8)))  float  v8f;

#define WMMA_BF16(A, B, C) \
  __builtin_amdgcn_wmma_f32_16x16x32_bf16(false, (A), false, (B), (short)0, (C), false, false)

// ---------------------------------------------------------------------------
// Kernel 0: weight prep. WT[n][k] = bf16(W[k][n]) so B-fragments are
// k-contiguous per lane (two b128 loads). WrbfT is zero-padded to [16][64].
// ---------------------------------------------------------------------------
__global__ __launch_bounds__(256) void prep_kernel(
    const float* __restrict__ Wq, const float* __restrict__ Wk,
    const float* __restrict__ Wv, const float* __restrict__ Wo,
    const float* __restrict__ Wrbf,
    __bf16* __restrict__ WqT, __bf16* __restrict__ WkT,
    __bf16* __restrict__ WvT, __bf16* __restrict__ WoT,
    __bf16* __restrict__ WrbfT) {
  int i = blockIdx.x * 256 + threadIdx.x;
  if (i < HID * HID) {
    int n = i / HID, k = i % HID;
    WqT[n * HID + k] = (__bf16)Wq[k * HID + n];
    WkT[n * HID + k] = (__bf16)Wk[k * HID + n];
    WvT[n * HID + k] = (__bf16)Wv[k * HID + n];
    WoT[n * HID + k] = (__bf16)Wo[k * HID + n];
  }
  if (i < 16 * 64) {
    int n = i / 64, k = i % 64;
    WrbfT[i] = (n < NHEADS && k < RBFD) ? (__bf16)Wrbf[k * NHEADS + n] : (__bf16)0.0f;
  }
}

// ---------------------------------------------------------------------------
// Kernel 1: Q/K/V projections. One wave per 16x16 output tile, K=256 -> 8 WMMA.
// Q gets 1/sqrt(HDIM) folded in. V is stored transposed: VT[d][p].
// ---------------------------------------------------------------------------
__global__ __launch_bounds__(32) void qkv_kernel(
    const float* __restrict__ lig, const float* __restrict__ prot,
    const __bf16* __restrict__ WqT, const __bf16* __restrict__ WkT,
    const __bf16* __restrict__ WvT,
    const float* __restrict__ bq, const float* __restrict__ bk,
    const float* __restrict__ bv,
    __bf16* __restrict__ Qbf, __bf16* __restrict__ Kbf,
    __bf16* __restrict__ VT) {
  int b = blockIdx.x;
  const float* X;
  const __bf16* WT;
  const float* bias;
  int which, tile;
  if (b < (LSZ / 16) * (HID / 16)) {
    which = 0; tile = b; X = lig; WT = WqT; bias = bq;
  } else if (b < 1024 + (PSZ / 16) * (HID / 16)) {
    which = 1; tile = b - 1024; X = prot; WT = WkT; bias = bk;
  } else {
    which = 2; tile = b - 3072; X = prot; WT = WvT; bias = bv;
  }
  int m0 = (tile / (HID / 16)) * 16;
  int n0 = (tile % (HID / 16)) * 16;
  int lane = threadIdx.x;
  int lp = lane & 15, half = lane >> 4;

  v8f c = {};
  for (int k0 = 0; k0 < HID; k0 += 32) {
    v16bf a, bm;
    const float* ap = X + (size_t)(m0 + lp) * HID + k0 + half * 8;
#pragma unroll
    for (int j = 0; j < 8; j++) { a[j] = (__bf16)ap[j]; a[8 + j] = (__bf16)ap[16 + j]; }
    const __bf16* bp = WT + (size_t)(n0 + lp) * HID + k0 + half * 8;
#pragma unroll
    for (int j = 0; j < 8; j++) { bm[j] = bp[j]; bm[8 + j] = bp[16 + j]; }
    c = WMMA_BF16(a, bm, c);
  }
  const float invs = 0.17677669529663687f;  // 1/sqrt(32)
#pragma unroll
  for (int r = 0; r < 8; r++) {
    int mr = m0 + r + half * 8;
    int nc = n0 + lp;
    float v = c[r] + bias[nc];
    if (which == 0)      Qbf[(size_t)mr * HID + nc] = (__bf16)(v * invs);
    else if (which == 1) Kbf[(size_t)mr * HID + nc] = (__bf16)v;
    else                 VT[(size_t)nc * PSZ + mr]  = (__bf16)v;   // VT[d][p]
  }
}

// ---------------------------------------------------------------------------
// Kernel 2: fused raw scores = QK^T (scaled) + rbf@Wrbf + brbf per 16Lx16P
// tile. rbf streamed from HBM exactly once. Scores staged in LDS and written
// out as coalesced b128 stores (each ligand row is 512B contiguous in
// [L][P][H]).
// ---------------------------------------------------------------------------
__global__ __launch_bounds__(32) void scores_kernel(
    const __bf16* __restrict__ Qbf, const __bf16* __restrict__ Kbf,
    const float* __restrict__ rbf, const __bf16* __restrict__ WrbfT,
    const float* __restrict__ brbf, float* __restrict__ attn) {
  __shared__ float bias_lds[16][16][NHEADS];
  __shared__ __align__(16) float sout[16 * 132];  // 128 payload + 4 pad per l-row
  int blk = blockIdx.x;
  int l0 = (blk >> 7) * 16;
  int p0 = (blk & 127) * 16;
  int lane = threadIdx.x;
  int lp = lane & 15, half = lane >> 4;

  v16bf wb0, wb1;
  {
    const __bf16* bp = WrbfT + lp * 64 + half * 8;
#pragma unroll
    for (int j = 0; j < 8; j++) { wb0[j] = bp[j]; wb0[8 + j] = bp[16 + j]; wb1[j] = bp[32 + j]; wb1[8 + j] = bp[48 + j]; }
  }

  // Distance bias GEMM: A rows = 16 p's of ligand row t, K padded 50 -> 64.
  for (int t = 0; t < 16; t++) {
    const float* rp = rbf + ((size_t)(l0 + t) * PSZ + (p0 + lp)) * RBFD;
    v16bf a0, a1;
#pragma unroll
    for (int j = 0; j < 8; j++) {
      a0[j]     = (__bf16)rp[half * 8 + j];
      a0[8 + j] = (__bf16)rp[16 + half * 8 + j];
      int k1 = 32 + half * 8 + j;
      a1[j] = (__bf16)rp[k1];
      int k2 = 48 + half * 8 + j;
      a1[8 + j] = (k2 < RBFD) ? (__bf16)rp[k2] : (__bf16)0.0f;
    }
    v8f cb = {};
    cb = WMMA_BF16(a0, wb0, cb);
    cb = WMMA_BF16(a1, wb1, cb);
    if (lp < NHEADS) {
#pragma unroll
      for (int r = 0; r < 8; r++) bias_lds[t][r + half * 8][lp] = cb[r];
    }
  }
  __syncthreads();

  // QK^T per head: K-depth = 32 -> one WMMA per head; stage into sout.
  for (int h = 0; h < NHEADS; h++) {
    v16bf a, bm;
    const __bf16* qp = Qbf + (size_t)(l0 + lp) * HID + h * HDIM + half * 8;
#pragma unroll
    for (int j = 0; j < 8; j++) { a[j] = qp[j]; a[8 + j] = qp[16 + j]; }
    const __bf16* kp = Kbf + (size_t)(p0 + lp) * HID + h * HDIM + half * 8;
#pragma unroll
    for (int j = 0; j < 8; j++) { bm[j] = kp[j]; bm[8 + j] = kp[16 + j]; }
    v8f c = {};
    c = WMMA_BF16(a, bm, c);
    float bb = brbf[h];
#pragma unroll
    for (int r = 0; r < 8; r++) {
      int lr = r + half * 8;
      sout[lr * 132 + lp * NHEADS + h] = c[r] + bias_lds[lr][lp][h] + bb;
    }
  }
  __syncthreads();

  // Coalesced writeback: 512 float4 chunks, 16 per lane.
  for (int cch = lane; cch < 512; cch += 32) {
    int l = cch >> 5;
    int o = (cch & 31) * 4;
    float4 v4 = *(const float4*)&sout[l * 132 + o];
    *(float4*)(attn + ((size_t)(l0 + l) * PSZ + p0) * NHEADS + o) = v4;
  }
}

// ---------------------------------------------------------------------------
// Kernel 3: per-(l,h) softmax stats (max, sum of exp). Reads are coalesced
// (h fastest across consecutive tids).
// ---------------------------------------------------------------------------
__global__ __launch_bounds__(256) void stats_kernel(
    const float* __restrict__ attn, float* __restrict__ stats) {
  int l = blockIdx.x;
  int tid = threadIdx.x;
  int h = tid & 7, ps = tid >> 3;
  const float* base = attn + (size_t)l * PSZ * NHEADS + h;
  __shared__ float red[NHEADS][32];
  __shared__ float mh[NHEADS];

  float m = -1e30f;
  for (int p = ps; p < PSZ; p += 32) m = fmaxf(m, base[(size_t)p * NHEADS]);
  red[h][ps] = m;
  __syncthreads();
  if (tid < NHEADS) {
    float mm = -1e30f;
    for (int i = 0; i < 32; i++) mm = fmaxf(mm, red[tid][i]);
    mh[tid] = mm;
  }
  __syncthreads();
  float mv = mh[h];
  float s = 0.0f;
  for (int p = ps; p < PSZ; p += 32) s += __expf(base[(size_t)p * NHEADS] - mv);
  __syncthreads();
  red[h][ps] = s;
  __syncthreads();
  if (tid < NHEADS) {
    float ss = 0.0f;
    for (int i = 0; i < 32; i++) ss += red[tid][i];
    stats[(l * NHEADS + tid) * 2 + 0] = mh[tid];
    stats[(l * NHEADS + tid) * 2 + 1] = ss;
  }
}

// ---------------------------------------------------------------------------
// Kernel 4: attend. One block per (16-ligand tile, P-chunk); 8 waves = 8
// heads. Raw score tiles stream HBM->LDS via global_load_async_to_lds_b128
// (double buffered, overlaps WMMA), normalized weights go back out via
// global_store_async_from_lds_b128 -- both fully coalesced 16B/lane.
// Partial attended sums per P-chunk keep the launch deterministic.
// ---------------------------------------------------------------------------
__global__ __launch_bounds__(256) void attend_kernel(
    float* __restrict__ attn, const float* __restrict__ stats,
    const __bf16* __restrict__ VT, float* __restrict__ att_part) {
  // 260-float rows (1040B = 65 banks): head-minor reads spread over 16 banks.
  __shared__ __align__(16) float inbuf[2][16 * 260];
  __shared__ __align__(16) float outbuf[16 * 260];
  int blk = blockIdx.x;
  int l0 = (blk / PCHUNKS) * 16;
  int pc = blk % PCHUNKS;
  int pbase = pc * PCH;
  int tid = threadIdx.x;
  int h = tid >> 5;            // wave == head
  int lane = tid & 31;
  int lp = lane & 15, half = lane >> 4;

  float m = stats[((l0 + lp) * NHEADS + h) * 2 + 0];
  float inv_s = 1.0f / stats[((l0 + lp) * NHEADS + h) * 2 + 1];

  auto issue_load = [&](int buf, int kt) {
    for (int c = tid; c < 1024; c += 256) {       // 1024 x 16B = 16KB tile
      int l = c >> 6;
      int o = (c & 63) * 4;
      unsigned lds = (unsigned)(uintptr_t)&inbuf[buf][l * 260 + o];
      const float* g =
          attn + ((size_t)(l0 + l) * PSZ + (pbase + kt * 32)) * NHEADS + o;
      asm volatile("global_load_async_to_lds_b128 %0, %1, off"
                   :: "v"(lds), "v"(g) : "memory");
    }
  };
  auto issue_store = [&](int kt) {
    for (int c = tid; c < 1024; c += 256) {
      int l = c >> 6;
      int o = (c & 63) * 4;
      unsigned lds = (unsigned)(uintptr_t)&outbuf[l * 260 + o];
      float* g = attn + ((size_t)(l0 + l) * PSZ + (pbase + kt * 32)) * NHEADS + o;
      asm volatile("global_store_async_from_lds_b128 %0, %1, off"
                   :: "v"(g), "v"(lds) : "memory");
    }
  };

  issue_load(0, 0);
  v8f c0 = {}, c1 = {};
  int cur = 0;
  for (int kt = 0; kt < PCH / 32; kt++) {
    // Each wave drains its own ASYNCcnt, then the barrier publishes all tiles
    // (and guarantees last iteration's outbuf stores have been consumed).
    asm volatile("s_wait_asynccnt 0x0" ::: "memory");
    __syncthreads();
    if (kt + 1 < PCH / 32) issue_load(cur ^ 1, kt + 1);

    v16bf a, b0, b1;
#pragma unroll
    for (int j = 0; j < 8; j++) {
      int p1 = half * 8 + j;
      float w1 = __expf(inbuf[cur][lp * 260 + p1 * NHEADS + h] - m) * inv_s;
      outbuf[lp * 260 + p1 * NHEADS + h] = w1;
      a[j] = (__bf16)w1;
      int p2 = 16 + half * 8 + j;
      float w2 = __expf(inbuf[cur][lp * 260 + p2 * NHEADS + h] - m) * inv_s;
      outbuf[lp * 260 + p2 * NHEADS + h] = w2;
      a[8 + j] = (__bf16)w2;
    }
    const __bf16* vp0 =
        VT + (size_t)(h * HDIM + lp) * PSZ + pbase + kt * 32 + half * 8;
    const __bf16* vp1 =
        VT + (size_t)(h * HDIM + 16 + lp) * PSZ + pbase + kt * 32 + half * 8;
#pragma unroll
    for (int j = 0; j < 8; j++) { b0[j] = vp0[j]; b0[8 + j] = vp0[16 + j]; b1[j] = vp1[j]; b1[8 + j] = vp1[16 + j]; }
    c0 = WMMA_BF16(a, b0, c0);
    c1 = WMMA_BF16(a, b1, c1);

    __syncthreads();           // outbuf complete across all waves
    issue_store(kt);
    cur ^= 1;
  }
  asm volatile("s_wait_asynccnt 0x0" ::: "memory");

  float* ap = att_part + (size_t)pc * LSZ * HID;
#pragma unroll
  for (int r = 0; r < 8; r++) {
    int lr = l0 + r + half * 8;
    ap[(size_t)lr * HID + h * HDIM + lp]      = c0[r];
    ap[(size_t)lr * HID + h * HDIM + 16 + lp] = c1[r];
  }
}

// ---------------------------------------------------------------------------
// Kernel 5: y = lig + (sum of attended partials) @ Wo + bo, then LayerNorm.
// ---------------------------------------------------------------------------
__global__ __launch_bounds__(256) void out_kernel(
    const float* __restrict__ att_part, const __bf16* __restrict__ WoT,
    const float* __restrict__ bo, const float* __restrict__ lig,
    const float* __restrict__ gamma, const float* __restrict__ beta,
    float* __restrict__ out) {
  __shared__ float y[16][HID];
  __shared__ float rs[16][16], rq[16][16];
  __shared__ float mu[16], rstd[16];
  int l0 = blockIdx.x * 16;
  int wave = threadIdx.x >> 5;
  int lane = threadIdx.x & 31;
  int lp = lane & 15, half = lane >> 4;

  for (int nt = wave * 2; nt < wave * 2 + 2; nt++) {
    int n0 = nt * 16;
    v8f c = {};
    for (int k0 = 0; k0 < HID; k0 += 32) {
      v16bf a, bm;
      const float* ap = att_part + (size_t)(l0 + lp) * HID + k0 + half * 8;
#pragma unroll
      for (int j = 0; j < 8; j++) {
        float s0 = ap[j]               + ap[(size_t)LSZ * HID + j] +
                   ap[2 * (size_t)LSZ * HID + j] + ap[3 * (size_t)LSZ * HID + j];
        float s1 = ap[16 + j]          + ap[(size_t)LSZ * HID + 16 + j] +
                   ap[2 * (size_t)LSZ * HID + 16 + j] + ap[3 * (size_t)LSZ * HID + 16 + j];
        a[j] = (__bf16)s0;
        a[8 + j] = (__bf16)s1;
      }
      const __bf16* bp = WoT + (size_t)(n0 + lp) * HID + k0 + half * 8;
#pragma unroll
      for (int j = 0; j < 8; j++) { bm[j] = bp[j]; bm[8 + j] = bp[16 + j]; }
      c = WMMA_BF16(a, bm, c);
    }
#pragma unroll
    for (int r = 0; r < 8; r++) {
      int lr = r + half * 8;
      int nc = n0 + lp;
      y[lr][nc] = c[r] + bo[nc] + lig[(size_t)(l0 + lr) * HID + nc];
    }
  }
  __syncthreads();

  int row = threadIdx.x >> 4;
  int sub = threadIdx.x & 15;
  float s = 0.0f, sq = 0.0f;
#pragma unroll
  for (int i = 0; i < 16; i++) {
    float v = y[row][sub * 16 + i];
    s += v; sq += v * v;
  }
  rs[row][sub] = s; rq[row][sub] = sq;
  __syncthreads();
  if (sub == 0) {
    float ts = 0.0f, tq = 0.0f;
    for (int i = 0; i < 16; i++) { ts += rs[row][i]; tq += rq[row][i]; }
    float mean = ts * (1.0f / HID);
    float var = tq * (1.0f / HID) - mean * mean;
    mu[row] = mean;
    rstd[row] = rsqrtf(var + 1e-5f);
  }
  __syncthreads();
  float mean = mu[row], rr = rstd[row];
#pragma unroll
  for (int i = 0; i < 16; i++) {
    int cidx = sub * 16 + i;
    out[(size_t)(l0 + row) * HID + cidx] =
        (y[row][cidx] - mean) * rr * gamma[cidx] + beta[cidx];
  }
}

// ---------------------------------------------------------------------------
extern "C" void kernel_launch(void* const* d_in, const int* in_sizes, int n_in,
                              void* d_out, int out_size, void* d_ws, size_t ws_size,
                              hipStream_t stream) {
  (void)in_sizes; (void)n_in; (void)out_size; (void)ws_size;
  const float* lig   = (const float*)d_in[0];
  const float* prot  = (const float*)d_in[1];
  const float* rbf   = (const float*)d_in[2];
  const float* Wq    = (const float*)d_in[5];
  const float* bq    = (const float*)d_in[6];
  const float* Wk    = (const float*)d_in[7];
  const float* bk    = (const float*)d_in[8];
  const float* Wv    = (const float*)d_in[9];
  const float* bv    = (const float*)d_in[10];
  const float* Wrbf  = (const float*)d_in[11];
  const float* brbf  = (const float*)d_in[12];
  const float* Wo    = (const float*)d_in[13];
  const float* bo    = (const float*)d_in[14];
  const float* gamma = (const float*)d_in[15];
  const float* beta  = (const float*)d_in[16];

  float* out_lig  = (float*)d_out;
  float* out_attn = out_lig + (size_t)LSZ * HID;  // raw scores -> final weights

  char* ws = (char*)d_ws;
  size_t off = 0;
  auto alloc = [&](size_t bytes) {
    void* p = ws + off;
    off += (bytes + 255) & ~(size_t)255;
    return p;
  };
  __bf16* WqT      = (__bf16*)alloc((size_t)HID * HID * 2);
  __bf16* WkT      = (__bf16*)alloc((size_t)HID * HID * 2);
  __bf16* WvT      = (__bf16*)alloc((size_t)HID * HID * 2);
  __bf16* WoT      = (__bf16*)alloc((size_t)HID * HID * 2);
  __bf16* WrbfT    = (__bf16*)alloc((size_t)16 * 64 * 2);
  __bf16* Qbf      = (__bf16*)alloc((size_t)LSZ * HID * 2);
  __bf16* Kbf      = (__bf16*)alloc((size_t)PSZ * HID * 2);
  __bf16* VT       = (__bf16*)alloc((size_t)HID * PSZ * 2);
  float*  stats    = (float*)alloc((size_t)LSZ * NHEADS * 2 * 4);
  float*  att_part = (float*)alloc((size_t)PCHUNKS * LSZ * HID * 4);

  prep_kernel<<<(HID * HID + 255) / 256, 256, 0, stream>>>(
      Wq, Wk, Wv, Wo, Wrbf, WqT, WkT, WvT, WoT, WrbfT);
  qkv_kernel<<<1024 + 2048 + 2048, 32, 0, stream>>>(
      lig, prot, WqT, WkT, WvT, bq, bk, bv, Qbf, Kbf, VT);
  scores_kernel<<<(LSZ / 16) * (PSZ / 16), 32, 0, stream>>>(
      Qbf, Kbf, rbf, WrbfT, brbf, out_attn);
  stats_kernel<<<LSZ, 256, 0, stream>>>(out_attn, stats);
  attend_kernel<<<(LSZ / 16) * PCHUNKS, 256, 0, stream>>>(
      out_attn, stats, VT, att_part);
  out_kernel<<<LSZ / 16, 256, 0, stream>>>(
      att_part, WoT, bo, lig, gamma, beta, out_lig);
}
